// GAT_FULL_PYG_38225208934547
// MI455X (gfx1250) — matrix operand order repaired and checked by reading, not compiled, not run on hardware
//
#include <hip/hip_runtime.h>
#include <hip/hip_bf16.h>

typedef __attribute__((ext_vector_type(16))) _Float16 v16h;
typedef __attribute__((ext_vector_type(8)))  float    v8f;

#define NNODES 100000
#define NEDGES 1600000
#define NEG_SLOPE 0.2f

// ---------- order-preserving f32 <-> u32 encoding for atomic max ----------
__device__ __forceinline__ unsigned enc_f32(float f) {
    unsigned u = __float_as_uint(f);
    return (u & 0x80000000u) ? ~u : (u | 0x80000000u);
}
__device__ __forceinline__ float dec_f32(unsigned u) {
    return __uint_as_float((u & 0x80000000u) ? (u ^ 0x80000000u) : ~u);
}
// enc(-inf) = ~0xFF800000 = 0x007FFFFF
#define ENC_NEG_INF 0x007FFFFFu

// =====================================================================
// GEMM: out[M,NCOLS] = A[M,K] @ W[K,NCOLS]  via v_wmma_f32_16x16x32_f16
// One wave computes a 16-row strip across all NPAD columns.
// W is staged transposed (n-major) in LDS as f16.
// =====================================================================
template<int K, int NCOLS, int NPAD>
__global__ __launch_bounds__(256) void gemm_wmma(const float* __restrict__ A,
                                                 const float* __restrict__ W,
                                                 float* __restrict__ out,
                                                 int M) {
    __shared__ __align__(32) _Float16 ldsW[NPAD * K];   // [n][k], f16
    const int tid = threadIdx.x;

    // cooperative stage: global W[k][n] f32  ->  LDS [n][k] f16 (pad cols = 0)
    for (int idx = tid; idx < NPAD * K; idx += 256) {
        int n = idx / K, k = idx - n * K;
        float v = (n < NCOLS) ? W[(size_t)k * NCOLS + n] : 0.0f;
        ldsW[idx] = (_Float16)v;
    }
    __syncthreads();

    const int wave = tid >> 5;
    const int lane = tid & 31;
    const int row0 = blockIdx.x * 128 + wave * 16;
    if (row0 >= M) return;

    const int mloc   = lane & 15;
    const int hiHalf = lane >> 4;          // 0: lanes 0-15, 1: lanes 16-31
    const int offA   = hiHalf * 16;        // A K-offset within a 32-chunk
    const int offB   = hiHalf * 16;        // B K-offset within a 32-chunk
    constexpr int NKC = K / 32;
    constexpr int NNT = NPAD / 16;

    v8f acc[NNT];
    v8f zero = {};
#pragma unroll
    for (int t = 0; t < NNT; ++t) acc[t] = zero;

    const int row = row0 + mloc;
    const bool rowOK = (row < M);
    const float* Arow = A + (size_t)row * K;

#pragma unroll
    for (int kc = 0; kc < NKC; ++kc) {
        const int kb = kc << 5;
        v16h a;
        if (rowOK) {
            // lane<16: K runs [kb+0..15]; lane>=16: [kb+16..31]
            const float4* p0 = (const float4*)(Arow + kb + offA);
            const float4* p1 = (const float4*)(Arow + kb + offA + 8);
            float4 f0 = p0[0], f1 = p0[1], f2 = p1[0], f3 = p1[1];
            a[0]=(_Float16)f0.x; a[1]=(_Float16)f0.y; a[2]=(_Float16)f0.z; a[3]=(_Float16)f0.w;
            a[4]=(_Float16)f1.x; a[5]=(_Float16)f1.y; a[6]=(_Float16)f1.z; a[7]=(_Float16)f1.w;
            a[8]=(_Float16)f2.x; a[9]=(_Float16)f2.y; a[10]=(_Float16)f2.z; a[11]=(_Float16)f2.w;
            a[12]=(_Float16)f3.x; a[13]=(_Float16)f3.y; a[14]=(_Float16)f3.z; a[15]=(_Float16)f3.w;
        } else {
#pragma unroll
            for (int i = 0; i < 16; ++i) a[i] = (_Float16)0.0f;
        }
#pragma unroll
        for (int nt = 0; nt < NNT; ++nt) {
            // B fragment: n = nt*16 + lane%16 ; k = kb + (lane<16 ? 0..15 : 16..31)
            const int n = (nt << 4) + mloc;
            v16h b = *(const v16h*)&ldsW[n * K + kb + offB];
            acc[nt] = __builtin_amdgcn_wmma_f32_16x16x32_f16(
                false, a, false, b, (short)0, acc[nt], false, false);
        }
    }

    // store: VGPR v -> row row0 + v + 8*hiHalf, col = nt*16 + lane%16
#pragma unroll
    for (int nt = 0; nt < NNT; ++nt) {
        const int col = (nt << 4) + mloc;
        if (col >= NCOLS) continue;
#pragma unroll
        for (int v = 0; v < 8; ++v) {
            const int r = row0 + v + hiHalf * 8;
            if (r < M) out[(size_t)r * NCOLS + col] = acc[nt][v];
        }
    }
}

// ---------- el/er = per-head dot(feat, al/ar) ----------
__global__ __launch_bounds__(256) void attn_proj(const float* __restrict__ feat,
                                                 const float* __restrict__ al,
                                                 const float* __restrict__ ar,
                                                 float* __restrict__ el,
                                                 float* __restrict__ er,
                                                 int Nn, int H, int F) {
    int n = blockIdx.x * 256 + threadIdx.x;
    if (n >= Nn) return;
    const float* f = feat + (size_t)n * H * F;
    for (int h = 0; h < H; ++h) {
        float sl = 0.0f, sr = 0.0f;
        for (int j = 0; j < F; ++j) {
            float v = f[h * F + j];
            sl += v * al[h * F + j];
            sr += v * ar[h * F + j];
        }
        el[(size_t)n * H + h] = sl;
        er[(size_t)n * H + h] = sr;
    }
}

// ---------- zero/neg-inf init for per-node reduction buffers ----------
__global__ __launch_bounds__(256) void init_bufs(unsigned* __restrict__ emax,
                                                 float* __restrict__ denom,
                                                 float* __restrict__ rst,
                                                 int NH, int NHF) {
    int i = blockIdx.x * 256 + threadIdx.x;
    if (i < NH) { emax[i] = ENC_NEG_INF; denom[i] = 0.0f; }
    if (i < NHF) rst[i] = 0.0f;
}

// ---------- pass 1: segment max of leaky-relu edge logits ----------
__global__ __launch_bounds__(256) void edge_max(const long long* __restrict__ ei,
                                                const float* __restrict__ el,
                                                const float* __restrict__ er,
                                                unsigned* __restrict__ emax,
                                                int Ecnt, int H) {
    int e = blockIdx.x * 256 + threadIdx.x;
    if (e >= Ecnt) return;
    long long s = ei[e];
    long long d = ei[(size_t)Ecnt + e];
    for (int h = 0; h < H; ++h) {
        float x = el[(size_t)s * H + h] + er[(size_t)d * H + h];
        x = (x > 0.0f) ? x : NEG_SLOPE * x;
        atomicMax(&emax[(size_t)d * H + h], enc_f32(x));
    }
}

// ---------- pass 2: a = exp(e - max); denom += a; rst += a*feat[src] ----------
__global__ __launch_bounds__(256) void edge_accum(const long long* __restrict__ ei,
                                                  const float* __restrict__ el,
                                                  const float* __restrict__ er,
                                                  const unsigned* __restrict__ emax,
                                                  const float* __restrict__ feat,
                                                  float* __restrict__ denom,
                                                  float* __restrict__ rst,
                                                  int Ecnt, int H, int F) {
    int e = blockIdx.x * 256 + threadIdx.x;
    if (e >= Ecnt) return;
    long long s = ei[e];
    long long d = ei[(size_t)Ecnt + e];
    for (int h = 0; h < H; ++h) {
        float x = el[(size_t)s * H + h] + er[(size_t)d * H + h];
        x = (x > 0.0f) ? x : NEG_SLOPE * x;
        float a = __expf(x - dec_f32(emax[(size_t)d * H + h]));
        atomicAdd(&denom[(size_t)d * H + h], a);
        const float* fs = feat + ((size_t)s * H + h) * F;
        float* rd = rst + ((size_t)d * H + h) * F;
        for (int j = 0; j < F; ++j) atomicAdd(&rd[j], a * fs[j]);
    }
}

// ---------- layer-1 finalize: normalize, +bias, elu(elu(.)) ----------
__global__ __launch_bounds__(256) void finalize1(const float* __restrict__ rst,
                                                 const float* __restrict__ denom,
                                                 const float* __restrict__ b,
                                                 float* __restrict__ hidden,
                                                 int Nn, int H, int F) {
    int n = blockIdx.x * 256 + threadIdx.x;
    if (n >= Nn) return;
    for (int h = 0; h < H; ++h) {
        float dn = denom[(size_t)n * H + h];
        float inv = (dn > 0.0f) ? 1.0f / dn : 0.0f;
        for (int j = 0; j < F; ++j) {
            float v = rst[((size_t)n * H + h) * F + j] * inv + b[h * F + j];
            v = (v > 0.0f) ? v : expm1f(v);   // GATConv internal elu
            v = (v > 0.0f) ? v : expm1f(v);   // wrapper elu
            hidden[((size_t)n * H + h) * F + j] = v;
        }
    }
}

// ---------- layer-2 finalize: normalize + bias (mean over 1 head = identity) ----------
__global__ __launch_bounds__(256) void finalize2(const float* __restrict__ rst,
                                                 const float* __restrict__ denom,
                                                 const float* __restrict__ b,
                                                 float* __restrict__ out,
                                                 int Nn, int C) {
    int i = blockIdx.x * 256 + threadIdx.x;
    if (i >= Nn * C) return;
    int n = i / C, c = i - n * C;
    float dn = denom[n];
    float v = (dn > 0.0f) ? rst[i] / dn : 0.0f;
    out[i] = v + b[c];
}

extern "C" void kernel_launch(void* const* d_in, const int* in_sizes, int n_in,
                              void* d_out, int out_size, void* d_ws, size_t ws_size,
                              hipStream_t stream) {
    const float*     x   = (const float*)d_in[0];
    const long long* ei  = (const long long*)d_in[1];   // [2,E] int64
    const float*     W1  = (const float*)d_in[2];
    const float*     al1 = (const float*)d_in[3];
    const float*     ar1 = (const float*)d_in[4];
    const float*     b1  = (const float*)d_in[5];
    const float*     W2  = (const float*)d_in[6];
    const float*     al2 = (const float*)d_in[7];
    const float*     ar2 = (const float*)d_in[8];
    const float*     b2  = (const float*)d_in[9];
    float* out = (float*)d_out;

    const int N = NNODES, E = NEDGES;
    const int H = 8, F = 8, HF = 64, C = 40;

    // ---- workspace layout (bytes); layer-2 buffers alias layer-1 buffers ----
    char* ws = (char*)d_ws;
    float*    feat1  = (float*)(ws);                                // N*64 f32
    float*    rst1   = (float*)(ws + (size_t)N * 64 * 4);           // N*64 f32
    float*    hidden = (float*)(ws + (size_t)N * 64 * 4 * 2);       // N*64 f32
    char* tail = ws + (size_t)N * 64 * 4 * 3;
    float*    el1    = (float*)(tail);                              // N*8
    float*    er1    = (float*)(tail + (size_t)N * 8 * 4);          // N*8
    unsigned* emax1  = (unsigned*)(tail + (size_t)N * 8 * 4 * 2);   // N*8
    float*    denom1 = (float*)(tail + (size_t)N * 8 * 4 * 3);      // N*8
    // layer-2 aliases (all smaller than their layer-1 counterparts)
    float*    feat2  = feat1;      // N*40
    float*    rst2   = rst1;       // N*40
    float*    el2    = el1;        // N
    float*    er2    = er1;        // N
    unsigned* emax2  = emax1;      // N
    float*    denom2 = denom1;     // N

    const int nodeBlocks  = (N + 255) / 256;
    const int edgeBlocks  = (E + 255) / 256;
    const int stripBlocks = (N + 127) / 128;

    // ===================== Layer 1 =====================
    gemm_wmma<256, 64, 64><<<stripBlocks, 256, 0, stream>>>(x, W1, feat1, N);
    attn_proj<<<nodeBlocks, 256, 0, stream>>>(feat1, al1, ar1, el1, er1, N, H, F);
    init_bufs<<<(N * HF + 255) / 256, 256, 0, stream>>>(emax1, denom1, rst1, N * H, N * HF);
    edge_max<<<edgeBlocks, 256, 0, stream>>>(ei, el1, er1, emax1, E, H);
    edge_accum<<<edgeBlocks, 256, 0, stream>>>(ei, el1, er1, emax1, feat1, denom1, rst1, E, H, F);
    finalize1<<<nodeBlocks, 256, 0, stream>>>(rst1, denom1, b1, hidden, N, H, F);

    // ===================== Layer 2 =====================
    gemm_wmma<64, 40, 48><<<stripBlocks, 256, 0, stream>>>(hidden, W2, feat2, N);
    attn_proj<<<nodeBlocks, 256, 0, stream>>>(feat2, al2, ar2, el2, er2, N, 1, C);
    init_bufs<<<(N * C + 255) / 256, 256, 0, stream>>>(emax2, denom2, rst2, N, N * C);
    edge_max<<<edgeBlocks, 256, 0, stream>>>(ei, el2, er2, emax2, E, 1);
    edge_accum<<<edgeBlocks, 256, 0, stream>>>(ei, el2, er2, emax2, feat2, denom2, rst2, E, 1, C);
    finalize2<<<(N * C + 255) / 256, 256, 0, stream>>>(rst2, denom2, b2, out, N, C);
}
